// CharRNNEmbedding_30623116821127
// MI455X (gfx1250) — matrix operand order repaired and checked by reading, not compiled, not run on hardware
//
#include <hip/hip_runtime.h>
#include <hip/hip_bf16.h>
#include <stdint.h>

// ---------------------------------------------------------------------------
// CharRNNEmbedding on MI455X (gfx1250, wave32, WMMA bf16 16x16x32, f32 accum)
//
// Head uses only layer-1 fwd output at t=0 and bwd output at t=T-1; both are
// FIRST steps of zero-state scans -> layer 1 collapses to one GEMM+activation
// per direction (w_hh_l1* * h=0 vanishes). Layer 0 runs full 16-step scans.
//
// Round-5 change: accumulators start from WMMA's inline-0 C operand on the
// first k-iteration; biases are added as scalars in the activation phase.
// This removes the 32 hoisted bias VGPRs that were causing the last spills.
//
// Workspace (bytes), ~19.5 MB of d_ws required.
// ---------------------------------------------------------------------------

typedef __attribute__((ext_vector_type(16))) __bf16 v16bf;
typedef __attribute__((ext_vector_type(8)))  float  v8f;
typedef __attribute__((ext_vector_type(4)))  unsigned int u32x4;

union FragU { v16bf v; u32x4 q[2]; };

typedef __attribute__((address_space(1))) const u32x4 GQ4;  // global-AS 16B chunk

#define WMMA_BF16(A, B, C) \
  __builtin_amdgcn_wmma_f32_16x16x32_bf16(false, (A), false, (B), (short)0, (C), false, false)

__device__ __forceinline__ unsigned short f32_to_bf16_rn(float f) {
  union { float f; unsigned int u; } x;
  x.f = f;
  unsigned int u = x.u;
  unsigned int r = (u + 0x7FFFu + ((u >> 16) & 1u)) >> 16;  // round-nearest-even
  return (unsigned short)r;
}

__device__ __forceinline__ float sigm(float x) { return 1.0f / (1.0f + __expf(-x)); }
__device__ __forceinline__ float tanh_f(float x) {
  x = fminf(fmaxf(x, -15.0f), 15.0f);
  float e = __expf(2.0f * x);
  return (e - 1.0f) / (e + 1.0f);
}

__device__ __forceinline__ v8f v8f_zero() {
  v8f r;
#pragma unroll
  for (int i = 0; i < 8; ++i) r[i] = 0.0f;
  return r;
}

// A-fragment (16x32 bf16) from LDS. Lanes 0-15: K 0..7 / 16..23 of row M=lane;
// lanes 16-31: K 8..15 / 24..31 of row M=lane-16.
__device__ __forceinline__ v16bf lds_load_a(const char* base, int rowStride,
                                            int rowBase, int kb, int lane) {
  int r = rowBase + (lane & 15);
  int sel = (lane < 16) ? 0 : 8;
  const char* p = base + r * rowStride + (kb + sel) * 2;
  FragU t;
  t.q[0] = *(const u32x4*)(p);
  t.q[1] = *(const u32x4*)(p + 32);
  return t.v;
}

// B-fragment (32x16 bf16) from row-major W[outDim][K]; column c of W^T is row
// c of W so K is contiguous. Lanes 0-15: K 0..7 / 16..23 of col N=lane; 16-31: K 8..15 / 24..31.
// Explicit global address space -> global_load_b128 (not flat_load).
__device__ __forceinline__ v16bf glb_load_b(const unsigned short* W,
                                            int K, int col, int kb, int lane) {
  int c = col + (lane & 15);
  int k0 = kb + ((lane < 16) ? 0 : 8);
  GQ4* p = (GQ4*)(uintptr_t)(W + (size_t)c * K + k0);
  FragU t;
  t.q[0] = p[0];
  t.q[1] = p[2];
  return t.v;
}

// ---------------------------------------------------------------------------
__global__ void cvt_f32_bf16(const float* __restrict__ src,
                             unsigned short* __restrict__ dst, int n) {
  int i = (blockIdx.x * blockDim.x + threadIdx.x) * 4;
  if (i + 3 < n) {
    float4 v = *(const float4*)(src + i);
    ushort4 o;
    o.x = f32_to_bf16_rn(v.x); o.y = f32_to_bf16_rn(v.y);
    o.z = f32_to_bf16_rn(v.z); o.w = f32_to_bf16_rn(v.w);
    *(ushort4*)(dst + i) = o;
  } else {
    for (; i < n; ++i) dst[i] = f32_to_bf16_rn(src[i]);
  }
}

__global__ void embed_gather(const int* __restrict__ ids,
                             const float* __restrict__ emb,
                             unsigned short* __restrict__ xg) {
  int i = blockIdx.x * blockDim.x + threadIdx.x;  // one thread per (word, t)
  if (i >= 4096 * 16) return;
  int id = ids[i];
  const float4* src = (const float4*)(emb + (size_t)id * 64);
  ushort4* dst = (ushort4*)(xg + (size_t)i * 64);
#pragma unroll
  for (int j = 0; j < 16; ++j) {
    float4 v = src[j];
    ushort4 o;
    o.x = f32_to_bf16_rn(v.x); o.y = f32_to_bf16_rn(v.y);
    o.z = f32_to_bf16_rn(v.z); o.w = f32_to_bf16_rn(v.w);
    dst[j] = o;
  }
}

// ---------------------------------------------------------------------------
// Layer-0 scan, one direction per blockIdx.y. 16 waves x 64 words per WG.
// Wave w owns units [w*16, w*16+16) of ALL FOUR gates across 4 row-tiles;
// c-state lives in VGPRs across the 16 steps; h round-trips via LDS.
#define NT0 64
#define NT1 32
#define NTHREADS 512
#define XSTRIDE 144   // 64 bf16 + 16B pad (conflict-free ds_read_b128)
#define HSTRIDE 528   // 256 bf16 + 16B pad

__launch_bounds__(NTHREADS)
__global__ void lstm_l0(const unsigned short* __restrict__ xg,
                        const unsigned short* __restrict__ wih_f,
                        const unsigned short* __restrict__ wih_b,
                        const unsigned short* __restrict__ whh_f,
                        const unsigned short* __restrict__ whh_b,
                        const float* __restrict__ bias_f,
                        const float* __restrict__ bias_b,
                        unsigned short* __restrict__ x1f,
                        unsigned short* __restrict__ x1b) {
  __shared__ __align__(16) char xbuf[NT0 * XSTRIDE];
  __shared__ __align__(16) char hbuf[NT0 * HSTRIDE];

  const int tid = threadIdx.x;
  const int lane = tid & 31;
  const int wv = tid >> 5;           // 0..15
  const int dir = blockIdx.y;
  const int word_base = blockIdx.x * NT0;
  const int ucol = wv * 16;          // this wave's 16-unit slice

  const unsigned short* Wih = dir ? wih_b : wih_f;
  const unsigned short* Whh = dir ? whh_b : whh_f;
  const float* bias = dir ? bias_b : bias_f;

  for (int i = tid; i < NT0 * HSTRIDE / 4; i += NTHREADS) ((unsigned int*)hbuf)[i] = 0u;

  {  // stage x for step 0: 64 rows x 128B, 16B per thread
    int t0 = dir ? 15 : 0;
    int row = tid >> 3, seg = tid & 7;
    uint4 v = *(const uint4*)(xg + ((size_t)(word_base + row) * 16 + t0) * 64 + seg * 8);
    *(uint4*)(xbuf + row * XSTRIDE + seg * 16) = v;
  }

  float bias_r[4];  // scalar per-lane gate biases, added in activation phase
#pragma unroll
  for (int g = 0; g < 4; ++g) bias_r[g] = bias[g * 256 + ucol + (lane & 15)];

  v8f cst[4];
#pragma unroll
  for (int rt = 0; rt < 4; ++rt) cst[rt] = v8f_zero();

  __syncthreads();

#pragma unroll 1
  for (int s = 0; s < 16; ++s) {
    // Opaque copies: stop LICM from hoisting every weight fragment out of the
    // scan loop (would need ~320 VGPRs -> scratch spills).
    const unsigned short* WihP = Wih;
    const unsigned short* WhhP = Whh;
    asm volatile("" : "+s"(WihP), "+s"(WhhP));

    v8f acc[4][4];

#pragma unroll
    for (int ki = 0; ki < 2; ++ki) {  // x part: K=64; ki==0 starts from inline C=0
      v16bf a[4];
#pragma unroll
      for (int rt = 0; rt < 4; ++rt)
        a[rt] = lds_load_a(xbuf, XSTRIDE, rt * 16, ki * 32, lane);
#pragma unroll
      for (int g = 0; g < 4; ++g) {
        v16bf b = glb_load_b(WihP, 64, g * 256 + ucol, ki * 32, lane);
#pragma unroll
        for (int rt = 0; rt < 4; ++rt)
          acc[rt][g] = (ki == 0) ? WMMA_BF16(a[rt], b, v8f_zero())
                                 : WMMA_BF16(a[rt], b, acc[rt][g]);
      }
    }
#pragma unroll
    for (int ki = 0; ki < 8; ++ki) {  // h part: K=256
      v16bf a[4];
#pragma unroll
      for (int rt = 0; rt < 4; ++rt)
        a[rt] = lds_load_a(hbuf, HSTRIDE, rt * 16, ki * 32, lane);
#pragma unroll
      for (int g = 0; g < 4; ++g) {
        v16bf b = glb_load_b(WhhP, 256, g * 256 + ucol, ki * 32, lane);
#pragma unroll
        for (int rt = 0; rt < 4; ++rt) acc[rt][g] = WMMA_BF16(a[rt], b, acc[rt][g]);
      }
    }

    __syncthreads();  // all waves done reading xbuf/hbuf

    unsigned short* hdst = nullptr;
    const int halfOff = dir ? 256 : 0;
    if (s == 0)  hdst = dir ? x1b : x1f;   // hb0[15] / hf0[0]
    if (s == 15) hdst = dir ? x1f : x1b;   // hb0[0]  / hf0[15]

#pragma unroll
    for (int rt = 0; rt < 4; ++rt) {
      v8f gi = acc[rt][0], gf = acc[rt][1], gg = acc[rt][2], go = acc[rt][3];
      v8f c = cst[rt];
#pragma unroll
      for (int r = 0; r < 8; ++r) {
        float cn = sigm(gf[r] + bias_r[1]) * c[r] +
                   sigm(gi[r] + bias_r[0]) * tanh_f(gg[r] + bias_r[2]);
        float h = sigm(go[r] + bias_r[3]) * tanh_f(cn);
        c[r] = cn;
        int rowL = rt * 16 + r + ((lane & 16) ? 8 : 0);
        int unit = ucol + (lane & 15);
        unsigned short hb = f32_to_bf16_rn(h);
        *(unsigned short*)(hbuf + rowL * HSTRIDE + unit * 2) = hb;
        if (hdst) hdst[(size_t)(word_base + rowL) * 512 + halfOff + unit] = hb;
      }
      cst[rt] = c;
    }

    if (s < 15) {  // stage x for next step
      int tn = dir ? (14 - s) : (s + 1);
      int row = tid >> 3, seg = tid & 7;
      uint4 v = *(const uint4*)(xg + ((size_t)(word_base + row) * 16 + tn) * 64 + seg * 8);
      *(uint4*)(xbuf + row * XSTRIDE + seg * 16) = v;
    }
    __syncthreads();
  }
}

// ---------------------------------------------------------------------------
#define ASTRIDE 1040  // 512 bf16 + 16B pad

__launch_bounds__(NTHREADS)
__global__ void lstm_l1_out(const unsigned short* __restrict__ x1f,
                            const unsigned short* __restrict__ x1b,
                            const unsigned short* __restrict__ wih1f,
                            const unsigned short* __restrict__ wih1b,
                            const float* __restrict__ b1f,
                            const float* __restrict__ b1b,
                            const unsigned short* __restrict__ wout,
                            const float* __restrict__ bout,
                            float* __restrict__ out) {
  __shared__ __align__(16) char abuf[NT1 * ASTRIDE];
  const int tid = threadIdx.x;
  const int lane = tid & 31;
  const int wv = tid >> 5;
  const int word_base = blockIdx.x * NT1;
  const int ucol = wv * 16;

  v8f hres[2][2];  // [pass][rt]

#pragma unroll 1
  for (int pass = 0; pass < 2; ++pass) {
    const unsigned short* X = pass ? x1b : x1f;
    const unsigned short* W = pass ? wih1b : wih1f;
    const float* B = pass ? b1b : b1f;

    {  // stage A tile [32, 512] bf16: 32 rows x 1024B, 64B per thread
      int row = tid >> 4, seg = tid & 15;
#pragma unroll
      for (int j = 0; j < 4; ++j) {
        uint4 v = *(const uint4*)(X + (size_t)(word_base + row) * 512 + seg * 32 + j * 8);
        *(uint4*)(abuf + row * ASTRIDE + seg * 64 + j * 16) = v;
      }
    }
    __syncthreads();

    float br[4];
#pragma unroll
    for (int g = 0; g < 4; ++g) br[g] = B[g * 256 + ucol + (lane & 15)];

    v8f acc[2][4];
#pragma unroll
    for (int ki = 0; ki < 16; ++ki) {  // ki==0 starts from inline C=0
      v16bf a[2];
#pragma unroll
      for (int rt = 0; rt < 2; ++rt)
        a[rt] = lds_load_a(abuf, ASTRIDE, rt * 16, ki * 32, lane);
#pragma unroll
      for (int g = 0; g < 4; ++g) {
        v16bf b = glb_load_b(W, 512, g * 256 + ucol, ki * 32, lane);
#pragma unroll
        for (int rt = 0; rt < 2; ++rt)
          acc[rt][g] = (ki == 0) ? WMMA_BF16(a[rt], b, v8f_zero())
                                 : WMMA_BF16(a[rt], b, acc[rt][g]);
      }
    }

#pragma unroll
    for (int rt = 0; rt < 2; ++rt) {
      v8f gi = acc[rt][0], gg = acc[rt][2], go = acc[rt][3];
      v8f h;
#pragma unroll
      for (int r = 0; r < 8; ++r) {
        float cn = sigm(gi[r] + br[0]) * tanh_f(gg[r] + br[2]);  // c_old = 0
        h[r] = sigm(go[r] + br[3]) * tanh_f(cn);
      }
      hres[pass][rt] = h;
    }
    __syncthreads();  // done reading abuf before reuse
  }

  // merged [32, 512] = h1_fwd || h1_bwd -> abuf (bf16)
#pragma unroll
  for (int rt = 0; rt < 2; ++rt)
#pragma unroll
    for (int r = 0; r < 8; ++r) {
      int rowL = rt * 16 + r + ((lane & 16) ? 8 : 0);
      int unit = ucol + (lane & 15);
      *(unsigned short*)(abuf + rowL * ASTRIDE + unit * 2) =
          f32_to_bf16_rn(hres[0][rt][r]);
      *(unsigned short*)(abuf + rowL * ASTRIDE + (256 + unit) * 2) =
          f32_to_bf16_rn(hres[1][rt][r]);
    }
  __syncthreads();

  // output projection: [32,512] @ [512,256] + b_out; wave w -> cols [w*16, w*16+16)
  float bo = bout[ucol + (lane & 15)];
  v8f acc3[2];
#pragma unroll
  for (int ki = 0; ki < 16; ++ki) {  // ki==0 starts from inline C=0
    v16bf b = glb_load_b(wout, 512, ucol, ki * 32, lane);
#pragma unroll
    for (int rt = 0; rt < 2; ++rt) {
      v16bf a = lds_load_a(abuf, ASTRIDE, rt * 16, ki * 32, lane);
      acc3[rt] = (ki == 0) ? WMMA_BF16(a, b, v8f_zero())
                           : WMMA_BF16(a, b, acc3[rt]);
    }
  }

#pragma unroll
  for (int rt = 0; rt < 2; ++rt)
#pragma unroll
    for (int r = 0; r < 8; ++r) {
      int rowL = rt * 16 + r + ((lane & 16) ? 8 : 0);
      int col = ucol + (lane & 15);
      out[(size_t)(word_base + rowL) * 256 + col] = acc3[rt][r] + bo;
    }
}

// ---------------------------------------------------------------------------
extern "C" void kernel_launch(void* const* d_in, const int* in_sizes, int n_in,
                              void* d_out, int out_size, void* d_ws, size_t ws_size,
                              hipStream_t stream) {
  (void)in_sizes; (void)n_in; (void)out_size; (void)ws_size;

  const int*   char_ids = (const int*)d_in[0];
  const float* char_emb = (const float*)d_in[1];
  const float* w_ih_l0f = (const float*)d_in[2];
  const float* w_hh_l0f = (const float*)d_in[3];
  const float* b_l0f    = (const float*)d_in[4];
  const float* w_ih_l0b = (const float*)d_in[5];
  const float* w_hh_l0b = (const float*)d_in[6];
  const float* b_l0b    = (const float*)d_in[7];
  const float* w_ih_l1f = (const float*)d_in[8];
  // d_in[9]  = w_hh_l1f (unused: h=0)
  const float* b_l1f    = (const float*)d_in[10];
  const float* w_ih_l1b = (const float*)d_in[11];
  // d_in[12] = w_hh_l1b (unused: h=0)
  const float* b_l1b    = (const float*)d_in[13];
  const float* w_out    = (const float*)d_in[14];
  const float* b_out    = (const float*)d_in[15];

  char* ws = (char*)d_ws;
  unsigned short* xg    = (unsigned short*)(ws + 0);
  unsigned short* wih0f = (unsigned short*)(ws + 8388608);
  unsigned short* wih0b = (unsigned short*)(ws + 8519680);
  unsigned short* whh0f = (unsigned short*)(ws + 8650752);
  unsigned short* whh0b = (unsigned short*)(ws + 9175040);
  unsigned short* wih1f = (unsigned short*)(ws + 9699328);
  unsigned short* wih1b = (unsigned short*)(ws + 10747904);
  unsigned short* woutb = (unsigned short*)(ws + 11796480);
  unsigned short* x1f   = (unsigned short*)(ws + 12058624);
  unsigned short* x1b   = (unsigned short*)(ws + 16252928);

  auto cvt = [&](const float* s, unsigned short* d, int n) {
    cvt_f32_bf16<<<(n / 4 + 255) / 256, 256, 0, stream>>>(s, d, n);
  };
  cvt(w_ih_l0f, wih0f, 1024 * 64);
  cvt(w_ih_l0b, wih0b, 1024 * 64);
  cvt(w_hh_l0f, whh0f, 1024 * 256);
  cvt(w_hh_l0b, whh0b, 1024 * 256);
  cvt(w_ih_l1f, wih1f, 1024 * 512);
  cvt(w_ih_l1b, wih1b, 1024 * 512);
  cvt(w_out,    woutb, 256 * 512);

  embed_gather<<<(4096 * 16 + 255) / 256, 256, 0, stream>>>(char_ids, char_emb, xg);

  lstm_l0<<<dim3(4096 / NT0, 2), NTHREADS, 0, stream>>>(xg, wih0f, wih0b, whh0f, whh0b,
                                                        b_l0f, b_l0b, x1f, x1b);

  lstm_l1_out<<<4096 / NT1, NTHREADS, 0, stream>>>(x1f, x1b, wih1f, wih1b, b_l1f, b_l1b,
                                                   woutb, b_out, (float*)d_out);
}
